// LocalAttention_49787260895467
// MI455X (gfx1250) — matrix operand order repaired and checked by reading, not compiled, not run on hardware
//
#include <hip/hip_runtime.h>
#include <math.h>

// ---------------------------------------------------------------------------
// LocalAttention for MI455X (gfx1250, wave32, WMMA 16x16x32 f16 -> f32 acc)
//
// out  = (ctx @ wo.T + bo)                      -> d_out[0 .. 4096*512)
// attn = softmax(band-masked QK^T/8)            -> d_out[4096*512 ..) (8x4096x4096)
//
// Dominant cost: writing the 536MB attn plane (~23us at 23.3 TB/s). That
// stream is ~2.8x the 192MB L2, so all attn-plane stores use non-temporal
// hints (ISA 7.3 TH=NT) to avoid sweeping L2 and evicting the f16 staging
// tiles. All matmuls use v_wmma_f32_16x16x32_f16 per CDNA5 ISA 7.12 layouts.
// ---------------------------------------------------------------------------

typedef __attribute__((ext_vector_type(16))) _Float16 v16h;
typedef __attribute__((ext_vector_type(8)))  _Float16 v8h;
typedef __attribute__((ext_vector_type(4)))  _Float16 v4h;
typedef __attribute__((ext_vector_type(8)))  float    v8f;
typedef __attribute__((ext_vector_type(4)))  float    v4f;

#define SEQL   4096
#define DMODEL 512
#define NHEADS 8
#define DK     64
#define WIN    128
#define NEGBIG (-1.0e9f)

// ---- A-fragment (16x32 f16, MxK). Lane holds row M=lane&15.
// elem e<8 : K = k0 + (lane>>4)*8 + e ; elem e>=8 : K = k0 + 16 + (lane>>4)*8 + (e-8)
__device__ __forceinline__ v16h load_A_f16(const _Float16* __restrict__ base, int ld,
                                           int row0, int k0) {
  const int lane = threadIdx.x & 31;
  const int r    = row0 + (lane & 15);
  const int kb   = k0 + ((lane >> 4) << 3);
  const _Float16* p = base + (size_t)r * ld + kb;
  const v8h lo = *(const v8h*)(p);        // K = kb..kb+7   (b128)
  const v8h hi = *(const v8h*)(p + 16);   // K = kb+16..kb+23 (b128)
  v16h a;
#pragma unroll
  for (int i = 0; i < 8; ++i) { a[i] = lo[i]; a[8 + i] = hi[i]; }
  return a;
}

// ---- B-fragment (32x16 f16, KxN) from a transposed source bt[n][k] (contiguous k).
// Lane holds col N=lane&15; elem e<8 : K = k0 + (lane>>4)*8 + e ; elem e>=8 : K = k0 + 16 + (lane>>4)*8 + (e-8)
__device__ __forceinline__ v16h load_B_f16(const _Float16* __restrict__ bt, int ld,
                                           int n0, int k0) {
  const int lane = threadIdx.x & 31;
  const int n    = n0 + (lane & 15);
  const int kb   = k0 + ((lane >> 4) << 3);
  const _Float16* p = bt + (size_t)n * ld + kb;
  const v8h lo = *(const v8h*)(p);        // K = kb..kb+7
  const v8h hi = *(const v8h*)(p + 16);   // K = kb+16..kb+23
  v16h b;
#pragma unroll
  for (int i = 0; i < 8; ++i) { b[i] = lo[i]; b[8 + i] = hi[i]; }
  return b;
}

// ---- f32 -> f16 convert (b128 loads, b64 stores)
__global__ void cvt_f32_f16(const float* __restrict__ src, _Float16* __restrict__ dst, int n4) {
  int i = blockIdx.x * blockDim.x + threadIdx.x;
  const int stride = gridDim.x * blockDim.x;
  for (; i < n4; i += stride) {
    const v4f x = ((const v4f*)src)[i];
    v4h y;
#pragma unroll
    for (int j = 0; j < 4; ++j) y[j] = (_Float16)x[j];
    ((v4h*)dst)[i] = y;
  }
}

// ---- zero attn plane (b128 NT stores, bypass L2 residency) + zero vT pad tail
__global__ void fill_zero(float* __restrict__ attn, size_t n4,
                          _Float16* __restrict__ pad, int npad) {
  const size_t i = (size_t)blockIdx.x * blockDim.x + threadIdx.x;
  const size_t stride = (size_t)gridDim.x * blockDim.x;
  const v4f z = {0.f, 0.f, 0.f, 0.f};
  for (size_t j = i; j < n4; j += stride)
    __builtin_nontemporal_store(z, (v4f*)attn + j);
  if (i < (size_t)npad) pad[i] = (_Float16)0.f;
}

// ---- GEMM: C(4096x512) = A(4096x512,f16) @ BT^T + bias. One wave = one 16x16 tile.
// mode 0/1: store f16 row-major [s][512] (Q/K)
// mode 2  : store f16 transposed vT[n*4096+s] (V)
// mode 3  : store f32 row-major to final output
__global__ void __launch_bounds__(128)
gemm_wmma(const _Float16* __restrict__ A, const _Float16* __restrict__ BT,
          const float* __restrict__ bias, _Float16* __restrict__ outh,
          float* __restrict__ outf, int mode) {
  const int nt = blockIdx.x * 4 + (threadIdx.x >> 5);  // 0..31 (N tiles of 16)
  const int mt = blockIdx.y;                           // 0..255 (M tiles of 16)
  const int lane = threadIdx.x & 31;

  v8f c = {};
  for (int kk = 0; kk < DMODEL; kk += 32) {
    if (kk + 32 < DMODEL)
      __builtin_prefetch(A + (size_t)(mt * 16 + (lane & 15)) * DMODEL + kk + 32, 0, 1);
    const v16h a = load_A_f16(A, DMODEL, mt * 16, kk);
    const v16h b = load_B_f16(BT, DMODEL, nt * 16, kk);
    c = __builtin_amdgcn_wmma_f32_16x16x32_f16(false, a, false, b, (short)0, c, false, false);
  }

  const int col = lane & 15;
  const int rb  = (lane >> 4) * 8;
  const int n   = nt * 16 + col;
  const float bv = bias[n];
#pragma unroll
  for (int i = 0; i < 8; ++i) {
    const int m = mt * 16 + rb + i;
    const float v = c[i] + bv;
    if (mode == 2)       outh[(size_t)n * SEQL + m]    = (_Float16)v;   // V transposed
    else if (mode == 3)  outf[(size_t)m * DMODEL + n]  = v;             // final f32
    else                 outh[(size_t)m * DMODEL + n]  = (_Float16)v;   // Q/K f16
  }
}

// ---- Banded attention: one wave per (16-query tile, head).
__global__ void __launch_bounds__(32)
attn_band(const _Float16* __restrict__ qh, const _Float16* __restrict__ kh,
          const _Float16* __restrict__ vT, float* __restrict__ attn_out,
          _Float16* __restrict__ ctxh) {
  __shared__ float sc[16][273];   // 17 key tiles max; stride 273 (odd) avoids bank conflicts

  const int qt   = blockIdx.x;            // 0..255
  const int h    = blockIdx.y;            // 0..7
  const int lane = threadIdx.x & 31;
  const int q0   = qt * 16;
  const int lo   = (q0 - WIN) > 0 ? (q0 - WIN) : 0;
  const int hi   = (q0 + 15 + WIN) < (SEQL - 1) ? (q0 + 15 + WIN) : (SEQL - 1);
  const int j0   = lo >> 4, j1 = hi >> 4;
  const int L    = (j1 - j0 + 1) * 16;    // band keys covered by this tile (<= 272)

  // Q fragments for d_k = 64 (two 32-K chunks), reused across all key tiles
  const v16h aq0 = load_A_f16(qh, DMODEL, q0, h * DK);
  const v16h aq1 = load_A_f16(qh, DMODEL, q0, h * DK + 32);

  const int col = lane & 15;
  const int rb  = (lane >> 4) * 8;

  // ---- scores: S = (Q K^T) / 8, band-masked, into LDS
  for (int jt = j0; jt <= j1; ++jt) {
    const v16h b0 = load_B_f16(kh, DMODEL, jt * 16, h * DK);
    const v16h b1 = load_B_f16(kh, DMODEL, jt * 16, h * DK + 32);
    v8f s = {};
    s = __builtin_amdgcn_wmma_f32_16x16x32_f16(false, aq0, false, b0, (short)0, s, false, false);
    s = __builtin_amdgcn_wmma_f32_16x16x32_f16(false, aq1, false, b1, (short)0, s, false, false);
    const int off = (jt - j0) * 16;
#pragma unroll
    for (int i = 0; i < 8; ++i) {
      const int qi = q0 + rb + i;
      const int kj = jt * 16 + col;
      const int d  = qi - kj;
      sc[rb + i][off + col] = (d <= WIN && d >= -WIN) ? s[i] * 0.125f : NEGBIG;
    }
  }
  __syncthreads();

  // ---- per-row softmax over the band (lanes 0..15, one query row each)
  if (lane < 16) {
    float m = -3.0e38f;
    for (int i = 0; i < L; ++i) m = fmaxf(m, sc[lane][i]);
    float sum = 0.f;
    for (int i = 0; i < L; ++i) { const float e = __expf(sc[lane][i] - m); sc[lane][i] = e; sum += e; }
    const float inv = 1.0f / sum;
    for (int i = 0; i < L; ++i) sc[lane][i] *= inv;
  }
  __syncthreads();

  // ---- write band probabilities into the dense attn output (NT: write-once stream)
  for (int r = 0; r < 16; ++r) {
    float* dst = attn_out + (size_t)(h * SEQL + q0 + r) * SEQL + (size_t)j0 * 16;
    for (int i = lane; i < L; i += 32)
      __builtin_nontemporal_store(sc[r][i], dst + i);
  }

  // ---- ctx = P @ V : chunks of 32 keys, A from LDS probs (f16), B from vT[d][s]
  v8f cv[4] = {{}, {}, {}, {}};
  const int nchunk = (L + 31) >> 5;
  for (int cc = 0; cc < nchunk; ++cc) {
    const int co = cc * 32;
    v16h pa;
    const int row = lane & 15;
    const int kb  = (lane >> 4) << 3;
#pragma unroll
    for (int e = 0; e < 8; ++e) {
      const int ka = co + kb + e;
      const int kc = co + 16 + kb + e;
      pa[e]     = (ka < L) ? (_Float16)sc[row][ka] : (_Float16)0.f;
      pa[8 + e] = (kc < L) ? (_Float16)sc[row][kc] : (_Float16)0.f;
    }
    const int s0 = j0 * 16 + co;   // global key base of chunk (vT has pad tail for overrun)
#pragma unroll
    for (int nt = 0; nt < 4; ++nt) {
      const v16h vb = load_B_f16(vT, SEQL, h * DK + nt * 16, s0);
      cv[nt] = __builtin_amdgcn_wmma_f32_16x16x32_f16(false, pa, false, vb, (short)0, cv[nt], false, false);
    }
  }
#pragma unroll
  for (int nt = 0; nt < 4; ++nt)
#pragma unroll
    for (int i = 0; i < 8; ++i)
      ctxh[(size_t)(q0 + rb + i) * DMODEL + h * DK + nt * 16 + col] = (_Float16)cv[nt][i];
}

// ---------------------------------------------------------------------------
extern "C" void kernel_launch(void* const* d_in, const int* in_sizes, int n_in,
                              void* d_out, int out_size, void* d_ws, size_t ws_size,
                              hipStream_t stream) {
  (void)in_sizes; (void)n_in; (void)out_size; (void)ws_size;

  const float* x  = (const float*)d_in[0];
  const float* wq = (const float*)d_in[1];
  const float* bq = (const float*)d_in[2];
  const float* wk = (const float*)d_in[3];
  const float* bk = (const float*)d_in[4];
  const float* wv = (const float*)d_in[5];
  const float* bv = (const float*)d_in[6];
  const float* wo = (const float*)d_in[7];
  const float* bo = (const float*)d_in[8];

  float* out  = (float*)d_out;                       // 4096*512
  float* attn = out + (size_t)SEQL * DMODEL;         // 8*4096*4096

  // workspace layout (f16 staging, ~23 MB; vT last so its 64-elem pad is safe)
  char* ws = (char*)d_ws;
  _Float16* xh   = (_Float16*)(ws + 0);          // 4096*512
  _Float16* wqh  = (_Float16*)(ws + 4194304);    // 512*512
  _Float16* wkh  = (_Float16*)(ws + 4718592);
  _Float16* wvh  = (_Float16*)(ws + 5242880);
  _Float16* woh  = (_Float16*)(ws + 5767168);
  _Float16* qh   = (_Float16*)(ws + 6291456);    // 4096*512
  _Float16* kh   = (_Float16*)(ws + 10485760);   // 4096*512
  _Float16* ctxh = (_Float16*)(ws + 14680064);   // 4096*512
  _Float16* vT   = (_Float16*)(ws + 18874368);   // 512*4096 (+64 pad)

  // 1) zero the dense attn plane (536MB, NT stores) + vT pad tail
  fill_zero<<<2048, 256, 0, stream>>>(attn, (size_t)NHEADS * SEQL * SEQL / 4,
                                      vT + (size_t)DMODEL * SEQL, 64);

  // 2) f32 -> f16 staging
  cvt_f32_f16<<<512, 256, 0, stream>>>(x,  xh,  SEQL * DMODEL / 4);
  cvt_f32_f16<<<256, 256, 0, stream>>>(wq, wqh, DMODEL * DMODEL / 4);
  cvt_f32_f16<<<256, 256, 0, stream>>>(wk, wkh, DMODEL * DMODEL / 4);
  cvt_f32_f16<<<256, 256, 0, stream>>>(wv, wvh, DMODEL * DMODEL / 4);
  cvt_f32_f16<<<256, 256, 0, stream>>>(wo, woh, DMODEL * DMODEL / 4);

  // 3) Q/K/V projections (V stored transposed for contiguous PV B-fragments)
  gemm_wmma<<<dim3(8, 256), 128, 0, stream>>>(xh, wqh, bq, qh, nullptr, 0);
  gemm_wmma<<<dim3(8, 256), 128, 0, stream>>>(xh, wkh, bk, kh, nullptr, 1);
  gemm_wmma<<<dim3(8, 256), 128, 0, stream>>>(xh, wvh, bv, vT, nullptr, 2);

  // 4) banded attention: scores -> softmax -> attn band + ctx
  attn_band<<<dim3(256, 8), 32, 0, stream>>>(qh, kh, vT, attn, ctxh);

  // 5) output projection (f32 store to d_out)
  gemm_wmma<<<dim3(8, 256), 128, 0, stream>>>(ctxh, woh, bo, nullptr, out, 3);
}
